// GPTLanguageModel_558345749134
// MI455X (gfx1250) — matrix operand order repaired and checked by reading, not compiled, not run on hardware
//
#include <hip/hip_runtime.h>

// ---- problem constants (match reference) ----
constexpr int B = 4, T = 1024, C = 1024, H = 16, V = 32000, L = 6;
constexpr int Dh = C / H;          // 64
constexpr int MROWS = B * T;       // 4096

typedef __attribute__((ext_vector_type(16))) _Float16 v16h;
typedef __attribute__((ext_vector_type(8)))  float    v8f;

union Frag16 { uint4 u[2]; v16h v; };

#define DEVI __device__ __forceinline__

DEVI float warpRedSum(float v) {
#pragma unroll
  for (int off = 16; off; off >>= 1) v += __shfl_xor(v, off, 32);
  return v;
}
DEVI float warpRedMax(float v) {
#pragma unroll
  for (int off = 16; off; off >>= 1) v = fmaxf(v, __shfl_xor(v, off, 32));
  return v;
}

DEVI float blockRedSum(float v) {
  __shared__ float buf[8];
  int wid = threadIdx.x >> 5, lane = threadIdx.x & 31;
  v = warpRedSum(v);
  if (lane == 0) buf[wid] = v;
  __syncthreads();
  v = (threadIdx.x < 8) ? buf[threadIdx.x] : 0.f;
  if (wid == 0) {
#pragma unroll
    for (int off = 4; off; off >>= 1) v += __shfl_xor(v, off, 32);
    if (lane == 0) buf[0] = v;
  }
  __syncthreads();
  float r = buf[0];
  __syncthreads();
  return r;
}

DEVI float blockRedMax(float v) {
  __shared__ float buf[8];
  int wid = threadIdx.x >> 5, lane = threadIdx.x & 31;
  v = warpRedMax(v);
  if (lane == 0) buf[wid] = v;
  __syncthreads();
  v = (threadIdx.x < 8) ? buf[threadIdx.x] : -INFINITY;
  if (wid == 0) {
#pragma unroll
    for (int off = 4; off; off >>= 1) v = fmaxf(v, __shfl_xor(v, off, 32));
    if (lane == 0) buf[0] = v;
  }
  __syncthreads();
  float r = buf[0];
  __syncthreads();
  return r;
}

// ---------------- one-time weight convert+transpose: W[R,Cc] f32 -> WT[Cc,R] f16 ----------------
__global__ __launch_bounds__(256)
void wconv_kernel(const float* __restrict__ W, _Float16* __restrict__ WT, int R, int Cc) {
  __shared__ _Float16 tile[32][33];
  const size_t mat = (size_t)R * Cc;
  const float*   Wz  = W  + (size_t)blockIdx.z * mat;
  _Float16*      WTz = WT + (size_t)blockIdx.z * mat;
  const int n0 = blockIdx.x * 32, k0 = blockIdx.y * 32;
  const int r = threadIdx.x >> 3, g = (threadIdx.x & 7) * 4;
  float4 f = *(const float4*)(Wz + (size_t)(k0 + r) * Cc + n0 + g);
  tile[r][g + 0] = (_Float16)f.x;
  tile[r][g + 1] = (_Float16)f.y;
  tile[r][g + 2] = (_Float16)f.z;
  tile[r][g + 3] = (_Float16)f.w;
  __syncthreads();
  union { _Float16 h[4]; uint2 u; } o;
  o.h[0] = tile[g + 0][r];
  o.h[1] = tile[g + 1][r];
  o.h[2] = tile[g + 2][r];
  o.h[3] = tile[g + 3][r];
  *(uint2*)(WTz + (size_t)(n0 + r) * R + k0 + g) = o.u;
}

// ---------------- embedding: x = tok_emb[idx] + pos_emb ----------------
__global__ __launch_bounds__(256)
void embed_kernel(const int* __restrict__ idx, const float* __restrict__ tok,
                  const float* __restrict__ pos, float* __restrict__ x) {
  int bt = blockIdx.x;
  int t  = bt & (T - 1);
  int tk = idx[bt];
  for (int c = threadIdx.x; c < C; c += 256)
    x[(size_t)bt * C + c] = tok[(size_t)tk * C + c] + pos[(size_t)t * C + c];
}

// ---------------- layernorm -> f16 activations ----------------
__global__ __launch_bounds__(256)
void ln_kernel(const float* __restrict__ x, const float* __restrict__ g,
               const float* __restrict__ b, _Float16* __restrict__ y) {
  int row = blockIdx.x;
  const float* xr = x + (size_t)row * C;
  float s = 0.f, ss = 0.f;
  for (int c = threadIdx.x; c < C; c += 256) {
    float v = xr[c];
    s += v; ss += v * v;
  }
  s  = blockRedSum(s);
  ss = blockRedSum(ss);
  float mean = s * (1.0f / C);
  float var  = ss * (1.0f / C) - mean * mean;
  float rstd = rsqrtf(var + 1e-5f);
  _Float16* yr = y + (size_t)row * C;
  for (int c = threadIdx.x; c < C; c += 256) {
    float v = (xr[c] - mean) * rstd * g[c] + b[c];
    yr[c] = (_Float16)v;
  }
}

// ---------------- WMMA GEMM: D = A(f16,[M,K]) @ BT(f16,[N,K])^T ----------------
// 256 threads = 8 waves; block tile 64x128; each wave: 16-row strip x 64 cols.
// Both tiles staged with GLOBAL_LOAD_ASYNC_TO_LDS_B128, double-buffered:
// issue k+1 tile async -> compute k tile (ds_load_b128 + 4 WMMA/wave) -> wait+barrier.
template <bool BIAS, bool RELU, bool RES, bool OUTF16>
__global__ __launch_bounds__(256)
void gemm_wmma(const _Float16* __restrict__ A, const _Float16* __restrict__ BT,
               const float* __restrict__ bias, void* __restrict__ Dout,
               const float* __restrict__ Resid, int N, int K) {
  __shared__ alignas(16) _Float16 Al[2][64][40];     // [buf][row][K]   (padded)
  __shared__ alignas(16) _Float16 Btl[2][128][40];   // [buf][col][K]   (padded)

  const int tid  = threadIdx.x;
  const int wid  = tid >> 5, lane = tid & 31;
  const int half = lane >> 4, l15 = lane & 15;
  const int wr   = wid & 3;          // 4 row strips of 16
  const int wc   = wid >> 2;         // 2 col groups of 64
  const int bm   = blockIdx.y, bn = blockIdx.x;

  v8f acc[4] = {{}, {}, {}, {}};

  // fixed per-thread staging addresses: quad q of 8 f16 (16B) in a 32-wide K slab
  const int ar_ = tid >> 2, aq = tid & 3;        // A: 64 rows x 4 quads
  const _Float16* aptr  = A  + (size_t)(bm * 64 + ar_) * K + aq * 8;
  const _Float16* bptr0 = BT + (size_t)(bn * 128 + ar_) * K + aq * 8;        // cols 0..63
  const _Float16* bptr1 = BT + (size_t)(bn * 128 + 64 + ar_) * K + aq * 8;   // cols 64..127
  unsigned aldst[2], bldst0[2], bldst1[2];
#pragma unroll
  for (int sbuf = 0; sbuf < 2; ++sbuf) {
    aldst[sbuf]  = (unsigned)(uintptr_t)&Al[sbuf][ar_][aq * 8];
    bldst0[sbuf] = (unsigned)(uintptr_t)&Btl[sbuf][ar_][aq * 8];
    bldst1[sbuf] = (unsigned)(uintptr_t)&Btl[sbuf][64 + ar_][aq * 8];
  }

  auto stage = [&](int sbuf, int kb) {
    const _Float16* ga  = aptr + kb;
    const _Float16* gb0 = bptr0 + kb;
    const _Float16* gb1 = bptr1 + kb;
    asm volatile("global_load_async_to_lds_b128 %0, %1, off" :: "v"(aldst[sbuf]),  "v"(ga)  : "memory");
    asm volatile("global_load_async_to_lds_b128 %0, %1, off" :: "v"(bldst0[sbuf]), "v"(gb0) : "memory");
    asm volatile("global_load_async_to_lds_b128 %0, %1, off" :: "v"(bldst1[sbuf]), "v"(gb1) : "memory");
  };

  stage(0, 0);
  asm volatile("s_wait_asynccnt 0x0" ::: "memory");
  __syncthreads();

  int cur = 0;
  for (int kb = 0; kb < K; kb += 32) {
    if (kb + 32 < K) stage(cur ^ 1, kb + 32);   // prefetch next tile (async, overlapped)

    Frag16 a;
    const uint4* arow = (const uint4*)&Al[cur][wr * 16 + l15][0];
    a.u[0] = arow[half];        // K = half*8 + e        (e<8)
    a.u[1] = arow[2 + half];    // K = 8 + half*8 + e    (e>=8)
#pragma unroll
    for (int t4 = 0; t4 < 4; ++t4) {
      Frag16 b;
      const uint4* brow = (const uint4*)&Btl[cur][wc * 64 + t4 * 16 + l15][0];
      b.u[0] = brow[half];      // K = half*8 + e        (e<8)
      b.u[1] = brow[2 + half];  // K = 8 + half*8 + e    (e>=8)
      acc[t4] = __builtin_amdgcn_wmma_f32_16x16x32_f16(false, a.v, false, b.v,
                                                       (short)0, acc[t4], false, false);
    }
    asm volatile("s_wait_asynccnt 0x0" ::: "memory");
    __syncthreads();
    cur ^= 1;
  }

  const int row0 = bm * 64 + wr * 16 + half * 8;
#pragma unroll
  for (int t4 = 0; t4 < 4; ++t4) {
    int n0 = bn * 128 + wc * 64 + t4 * 16 + l15;
    float bb = BIAS ? bias[n0] : 0.f;
#pragma unroll
    for (int r = 0; r < 8; ++r) {
      size_t o = (size_t)(row0 + r) * N + n0;
      float v = acc[t4][r] + bb;
      if (RELU) v = fmaxf(v, 0.f);
      if (RES)  v += Resid[o];
      if (OUTF16) ((_Float16*)Dout)[o] = (_Float16)v;
      else        ((float*)Dout)[o] = v;
    }
  }
}

// ---------------- s/d projections: s[b,h,t] = h[b,t,h,:] . a_src[h,:] ----------------
__global__ __launch_bounds__(256)
void sd_kernel(const float* __restrict__ h, const float* __restrict__ asrc,
               const float* __restrict__ adst, float* __restrict__ s, float* __restrict__ d) {
  int wid = threadIdx.x >> 5, lane = threadIdx.x & 31;
  int gw = blockIdx.x * 8 + wid;       // (b, hh, t)
  int t  = gw & (T - 1);
  int hh = (gw >> 10) & (H - 1);
  int b  = gw >> 14;
  const float2* hb = (const float2*)(h + ((size_t)(b * T + t)) * C + hh * Dh);
  const float2* as = (const float2*)(asrc + hh * Dh);
  const float2* ad = (const float2*)(adst + hh * Dh);
  float2 hv = hb[lane];
  float2 av = as[lane];
  float2 dv2 = ad[lane];
  float sv = hv.x * av.x  + hv.y * av.y;
  float dv = hv.x * dv2.x + hv.y * dv2.y;
  sv = warpRedSum(sv);
  dv = warpRedSum(dv);
  if (lane == 0) {
    size_t o = ((size_t)(b * H + hh)) * T + t;
    s[o] = sv; d[o] = dv;
  }
}

// ---------------- causal GAT attention, online softmax, one wave per (b,h,j) ----------------
__global__ __launch_bounds__(256)
void attn_kernel(const float* __restrict__ h, const float* __restrict__ s,
                 const float* __restrict__ dvec, const float* __restrict__ bgl,
                 float* __restrict__ x) {
  const float LOG2 = 0.6931471805599453f;
  int wid = threadIdx.x >> 5, lane = threadIdx.x & 31;
  int gw = blockIdx.x * 8 + wid;       // (b, hh, j)
  int j  = gw & (T - 1);
  int hh = (gw >> 10) & (H - 1);
  int b  = gw >> 14;

  const float* sb = s + ((size_t)(b * H + hh)) * T;
  const float* hb = h + (size_t)b * T * C + hh * Dh;
  float dj = dvec[((size_t)(b * H + hh)) * T + j];

  float m = -INFINITY, lsum = 0.f;
  float acc0 = 0.f, acc1 = 0.f;      // channels 2*lane, 2*lane+1
  for (int c0 = 0; c0 <= j; c0 += 32) {
    int i = c0 + lane;
    float sc = -INFINITY;
    if (i <= j) {
      float v = sb[i] + dj;
      sc = (v >= 0.f) ? v : 0.2f * v;   // leaky_relu, slope 0.2
      if (i == j) sc += LOG2;           // duplicated self-loop -> 2*exp(score)
    }
    float cm   = warpRedMax(sc);
    float mnew = fmaxf(m, cm);
    float scl  = __expf(m - mnew);      // m=-inf on first pass -> 0
    lsum *= scl; acc0 *= scl; acc1 *= scl;
    m = mnew;
    float w = (i <= j) ? __expf(sc - mnew) : 0.f;
    lsum += w;
    const float* hp = hb + (size_t)c0 * C + 2 * lane;
#pragma unroll 4
    for (int ii = 0; ii < 32; ++ii) {
      float wv = __shfl(w, ii, 32);
      float2 hv = *(const float2*)(hp + (size_t)ii * C);
      acc0 = fmaf(wv, hv.x, acc0);
      acc1 = fmaf(wv, hv.y, acc1);
    }
  }
  float tot = warpRedSum(lsum);
  float inv = 1.0f / tot;
  const float2 bgv = *(const float2*)(bgl + hh * Dh + 2 * lane);
  float2* xp = (float2*)(x + ((size_t)(b * T + j)) * C + hh * Dh + 2 * lane);
  float2 xv = *xp;
  xv.x += acc0 * inv + bgv.x;
  xv.y += acc1 * inv + bgv.y;
  *xp = xv;
}

// ---------------- loss ----------------
__global__ void zero_loss_kernel(float* p) { *p = 0.f; }

__global__ __launch_bounds__(256)
void loss_kernel(const float* __restrict__ logits, const int* __restrict__ targets,
                 float* __restrict__ loss) {
  int row = blockIdx.x;
  const float* lr = logits + (size_t)row * V;
  float mx = -INFINITY;
  for (int v = threadIdx.x; v < V; v += 256) mx = fmaxf(mx, lr[v]);
  mx = blockRedMax(mx);
  float se = 0.f;
  for (int v = threadIdx.x; v < V; v += 256) se += __expf(lr[v] - mx);
  se = blockRedSum(se);
  if (threadIdx.x == 0) {
    float logZ = mx + __logf(se);
    float lt = lr[targets[row]];
    atomicAdd(loss, (logZ - lt) * (1.0f / (float)MROWS));
  }
}

// ---------------- driver ----------------
extern "C" void kernel_launch(void* const* d_in, const int* in_sizes, int n_in,
                              void* d_out, int out_size, void* d_ws, size_t ws_size,
                              hipStream_t stream) {
  (void)in_sizes; (void)n_in; (void)out_size; (void)ws_size;
  const int*   idx     = (const int*)d_in[0];
  const int*   targets = (const int*)d_in[1];
  // d_in[2] = edge_index (unused; causal mask is implicit)
  const float* tok_emb = (const float*)d_in[3];
  const float* pos_emb = (const float*)d_in[4];
  const float* Wg      = (const float*)d_in[5];
  const float* a_src   = (const float*)d_in[6];
  const float* a_dst   = (const float*)d_in[7];
  const float* bg      = (const float*)d_in[8];
  const float* ln1_g   = (const float*)d_in[9];
  const float* ln1_b   = (const float*)d_in[10];
  const float* ln2_g   = (const float*)d_in[11];
  const float* ln2_b   = (const float*)d_in[12];
  const float* W1      = (const float*)d_in[13];
  const float* b1      = (const float*)d_in[14];
  const float* W2      = (const float*)d_in[15];
  const float* b2      = (const float*)d_in[16];
  const float* lnf_g   = (const float*)d_in[17];
  const float* lnf_b   = (const float*)d_in[18];
  const float* Wlm     = (const float*)d_in[19];
  const float* blm     = (const float*)d_in[20];

  char* ws = (char*)d_ws;
  size_t off = 0;
  auto wsAlloc = [&](size_t bytes) -> void* {
    void* p = ws + off;
    off += (bytes + 255) & ~(size_t)255;
    return p;
  };
  float*    x    = (float*)wsAlloc((size_t)MROWS * C * 4);
  float*    h    = (float*)wsAlloc((size_t)MROWS * C * 4);
  _Float16* y    = (_Float16*)wsAlloc((size_t)MROWS * C * 2);
  _Float16* u    = (_Float16*)wsAlloc((size_t)MROWS * 4 * C * 2);
  float*    sv   = (float*)wsAlloc((size_t)B * H * T * 4);
  float*    dv   = (float*)wsAlloc((size_t)B * H * T * 4);
  _Float16* WgT  = (_Float16*)wsAlloc((size_t)L * C * C * 2);
  _Float16* W1T  = (_Float16*)wsAlloc((size_t)L * C * 4 * C * 2);
  _Float16* W2T  = (_Float16*)wsAlloc((size_t)L * 4 * C * C * 2);
  _Float16* WlmT = (_Float16*)wsAlloc((size_t)C * V * 2);

  float* logits = (float*)d_out;
  float* loss   = logits + (size_t)MROWS * V;

  // one-time (per launch) weight f32 -> f16 transpose: WT[N,K] from W[K,N]
  wconv_kernel<<<dim3(C / 32, C / 32, L), 256, 0, stream>>>(Wg, WgT, C, C);
  wconv_kernel<<<dim3(4 * C / 32, C / 32, L), 256, 0, stream>>>(W1, W1T, C, 4 * C);
  wconv_kernel<<<dim3(C / 32, 4 * C / 32, L), 256, 0, stream>>>(W2, W2T, 4 * C, C);
  wconv_kernel<<<dim3(V / 32, C / 32, 1), 256, 0, stream>>>(Wlm, WlmT, C, V);

  embed_kernel<<<MROWS, 256, 0, stream>>>(idx, tok_emb, pos_emb, x);

  for (int l = 0; l < L; ++l) {
    ln_kernel<<<MROWS, 256, 0, stream>>>(x, ln1_g + l * C, ln1_b + l * C, y);
    // h = LN(x) @ Wg[l]
    gemm_wmma<false, false, false, false><<<dim3(C / 128, MROWS / 64), 256, 0, stream>>>(
        y, WgT + (size_t)l * C * C, nullptr, h, nullptr, C, C);
    sd_kernel<<<(B * H * T) / 8, 256, 0, stream>>>(h, a_src + l * H * Dh, a_dst + l * H * Dh, sv, dv);
    attn_kernel<<<(B * H * T) / 8, 256, 0, stream>>>(h, sv, dv, bg + l * C, x);
    ln_kernel<<<MROWS, 256, 0, stream>>>(x, ln2_g + l * C, ln2_b + l * C, y);
    // u = relu(LN2(x) @ W1 + b1)   (f16 out)
    gemm_wmma<true, true, false, true><<<dim3(4 * C / 128, MROWS / 64), 256, 0, stream>>>(
        y, W1T + (size_t)l * C * 4 * C, b1 + l * 4 * C, u, nullptr, 4 * C, C);
    // x = x + u @ W2 + b2
    gemm_wmma<true, false, true, false><<<dim3(C / 128, MROWS / 64), 256, 0, stream>>>(
        u, W2T + (size_t)l * 4 * C * C, b2 + l * C, x, x, C, 4 * C);
  }

  ln_kernel<<<MROWS, 256, 0, stream>>>(x, lnf_g, lnf_b, y);
  // logits = LNf(x) @ Wlm + blm
  gemm_wmma<true, false, false, false><<<dim3(V / 128, MROWS / 64), 256, 0, stream>>>(
      y, WlmT, blm, logits, nullptr, V, C);

  zero_loss_kernel<<<1, 1, 0, stream>>>(loss);
  loss_kernel<<<MROWS, 256, 0, stream>>>(logits, targets, loss);
}
